// BidirectionalRNN_46566035423372
// MI455X (gfx1250) — matrix operand order repaired and checked by reading, not compiled, not run on hardware
//
#include <hip/hip_runtime.h>
#include <hip/hip_bf16.h>

typedef __bf16 bf16_t;
typedef __attribute__((ext_vector_type(16))) __bf16 v16bf;
typedef __attribute__((ext_vector_type(8)))  __bf16 v8bf;
typedef __attribute__((ext_vector_type(8)))  float  v8f;

static constexpr int B_  = 64;
static constexpr int T_  = 1024;
static constexpr int H_  = 512;
static constexpr int NIN = 512;
static constexpr int NBLK = 4;            // blocks per direction in the scan
static constexpr int LDSW_LD = 520;       // padded LDS row stride (bf16 elems)
static constexpr int LDS_BYTES = 128 * LDSW_LD * 2;  // 133,120 B < 320 KB/WGP

// ---------------------------------------------------------------------------
// A-operand loader: 16x32 bf16 tile, ISA 7.12.2 layout.
// lane L: row m = L&15; elems 0..7 -> K = 8*(L>>4)+0..7 ; elems 8..15 -> +16.
// ---------------------------------------------------------------------------
__device__ __forceinline__ v16bf load_a_tile(const bf16_t* __restrict__ base,
                                             int ld, int kk, int lane) {
  const int m = lane & 15, half = lane >> 4;
  const bf16_t* p = base + (size_t)m * ld + kk + half * 8;
  v8bf lo = *(const v8bf*)p;          // K = kk + half*8 .. +7
  v8bf hi = *(const v8bf*)(p + 16);   // K = kk + 16 + half*8 .. +7
  v16bf r;
#pragma unroll
  for (int i = 0; i < 8; ++i) { r[i] = lo[i]; r[i + 8] = hi[i]; }
  return r;
}

// ---------------------------------------------------------------------------
// B-operand loader from row-major weight W[n][k] (B[k][n] = W[n][k]):
// lane L: col n = L&15; elems 0..7 -> K = kk + 8*(L>>4)+0..7 ; elems 8..15 -> +16.
// ---------------------------------------------------------------------------
__device__ __forceinline__ v16bf load_b_tile(const bf16_t* __restrict__ base,
                                             int ld, int kk, int lane) {
  const int n = lane & 15, koff = (lane >> 4) * 8;
  const bf16_t* p = base + (size_t)n * ld + kk + koff;
  v8bf lo = *(const v8bf*)p;
  v8bf hi = *(const v8bf*)(p + 16);
  v16bf r;
#pragma unroll
  for (int i = 0; i < 8; ++i) { r[i] = lo[i]; r[i + 8] = hi[i]; }
  return r;
}

#define WMMA_BF16(A, Bm, C) \
  __builtin_amdgcn_wmma_f32_16x16x32_bf16(false, (A), false, (Bm), (short)0, (C), false, false)

// ---------------------------------------------------------------------------
// fp32 -> bf16 conversion (weights, activations)
// ---------------------------------------------------------------------------
__global__ void cvt_f32_to_bf16(const float* __restrict__ src,
                                bf16_t* __restrict__ dst, int n) {
  int i = blockIdx.x * blockDim.x + threadIdx.x;
  int stride = gridDim.x * blockDim.x;
  for (; i < n; i += stride) dst[i] = (bf16_t)src[i];
}

// zero h ping-pong buffers and barrier counters before each scan
__global__ void init_state(bf16_t* __restrict__ h, int nh,
                           unsigned* __restrict__ c, int nc) {
  int i = blockIdx.x * blockDim.x + threadIdx.x;
  int stride = gridDim.x * blockDim.x;
  for (int k = i; k < nh; k += stride) h[k] = (bf16_t)0.0f;
  for (int k = i; k < nc; k += stride) c[k] = 0u;
}

// ---------------------------------------------------------------------------
// pre[bt][j] = sum_k X[bt][k] * W[j][k] + (b_ih[j] + b_hh[j])
// Block = 256 thr (8 waves, 4x2), tile 128(M) x 64(N); wave tile 32x32 (4 WMMA).
// blockIdx.z selects direction (f/b).
// ---------------------------------------------------------------------------
__global__ void gemm_pre_kernel(const bf16_t* __restrict__ X, int K,
                                const bf16_t* __restrict__ Wf,
                                const bf16_t* __restrict__ Wb,
                                const float* __restrict__ bif, const float* __restrict__ bhf,
                                const float* __restrict__ bib, const float* __restrict__ bhb,
                                float* __restrict__ pref, float* __restrict__ preb) {
  const int dir = blockIdx.z;
  const bf16_t* W  = dir ? Wb  : Wf;
  const float*  bi = dir ? bib : bif;
  const float*  bh = dir ? bhb : bhf;
  float*        pre = dir ? preb : pref;

  const int tid = threadIdx.x, lane = tid & 31, w = tid >> 5;
  const int row0 = blockIdx.x * 128 + (w & 3) * 32;
  const int col0 = blockIdx.y * 64  + (w >> 2) * 32;

  const bf16_t* Xa0 = X + (size_t)row0 * K;
  const bf16_t* Xa1 = X + (size_t)(row0 + 16) * K;
  const bf16_t* Wb0 = W + (size_t)col0 * K;
  const bf16_t* Wb1 = W + (size_t)(col0 + 16) * K;

  v8f c00 = {}, c01 = {}, c10 = {}, c11 = {};
  for (int kk = 0; kk < K; kk += 32) {
    v16bf a0 = load_a_tile(Xa0, K, kk, lane);
    v16bf a1 = load_a_tile(Xa1, K, kk, lane);
    v16bf b0 = load_b_tile(Wb0, K, kk, lane);
    v16bf b1 = load_b_tile(Wb1, K, kk, lane);
    c00 = WMMA_BF16(a0, b0, c00);
    c01 = WMMA_BF16(a0, b1, c01);
    c10 = WMMA_BF16(a1, b0, c10);
    c11 = WMMA_BF16(a1, b1, c11);
  }

  const int half = lane >> 4, nl = lane & 15;
  const int n0 = col0 + nl, n1 = n0 + 16;
  const float bb0 = bi[n0] + bh[n0];
  const float bb1 = bi[n1] + bh[n1];
#pragma unroll
  for (int r = 0; r < 8; ++r) {
    const int m0 = row0 + r + 8 * half;
    const int m1 = m0 + 16;
    pre[(size_t)m0 * H_ + n0] = c00[r] + bb0;
    pre[(size_t)m0 * H_ + n1] = c01[r] + bb1;
    pre[(size_t)m1 * H_ + n0] = c10[r] + bb0;
    pre[(size_t)m1 * H_ + n1] = c11[r] + bb1;
  }
}

// ---------------------------------------------------------------------------
// Persistent recurrent scan. grid = (NBLK, 2 dirs), block = 256 thr (8 waves).
// Block owns 128 columns of w_hh, resident in LDS for all T steps.
// h ping-pongs through global (L2) bf16 buffers; per-direction global atomic
// barrier each step. Epilogue: + pre, tanh, emit bf16 (next layer) or f32 out.
// ---------------------------------------------------------------------------
__global__ void rnn_scan_kernel(const float* __restrict__ pre_f,
                                const float* __restrict__ pre_b,
                                const bf16_t* __restrict__ whh_f,
                                const bf16_t* __restrict__ whh_b,
                                bf16_t* __restrict__ hbuf,      // [2 dirs][2][B_*H_]
                                unsigned* __restrict__ ctr,     // [2]
                                bf16_t* __restrict__ out_bf16,  // layer-0 path
                                float* __restrict__ out_f32) {  // layer-1 path
  extern __shared__ bf16_t lds_w[];  // [128][LDSW_LD]
  const int dir = blockIdx.y;
  const float*  pre = dir ? pre_b : pre_f;
  const bf16_t* whh = dir ? whh_b : whh_f;
  const int j0 = blockIdx.x * 128;
  const int tid = threadIdx.x, lane = tid & 31, w = tid >> 5;

  // Stage this block's 128 w_hh rows (row-major, padded stride) once.
  for (int idx = tid; idx < 128 * 64; idx += 256) {
    const int r = idx >> 6, c8 = (idx & 63) << 3;
    *(v8bf*)&lds_w[r * LDSW_LD + c8] = *(const v8bf*)&whh[(size_t)(j0 + r) * H_ + c8];
  }
  __syncthreads();

  bf16_t* hbase = hbuf + (size_t)dir * 2 * B_ * H_;
  unsigned* myctr = ctr + dir;
  const int half = lane >> 4, nl = lane & 15;
  const int jcol = j0 + 16 * w + nl;
  const bf16_t* ldsb = lds_w + (16 * w + nl) * LDSW_LD;  // this lane's W row

  for (int s = 0; s < T_; ++s) {
    const int t = dir ? (T_ - 1 - s) : s;
    const bf16_t* hprev = hbase + (size_t)(s & 1) * B_ * H_;
    bf16_t* hnext = hbase + (size_t)((s + 1) & 1) * B_ * H_;

    v8f c0 = {}, c1 = {}, c2 = {}, c3 = {};
    for (int kk = 0; kk < H_; kk += 32) {
      // B operand: K = kk + 8*half + 0..7 and +16 from LDS (ds_load_b128 x2)
      const bf16_t* pb = ldsb + kk + half * 8;
      v8bf blo = *(const v8bf*)pb;
      v8bf bhi = *(const v8bf*)(pb + 16);
      v16bf bm;
#pragma unroll
      for (int i = 0; i < 8; ++i) { bm[i] = blo[i]; bm[i + 8] = bhi[i]; }

      v16bf a0 = load_a_tile(hprev +  0 * H_, H_, kk, lane);
      v16bf a1 = load_a_tile(hprev + 16 * H_, H_, kk, lane);
      v16bf a2 = load_a_tile(hprev + 32 * H_, H_, kk, lane);
      v16bf a3 = load_a_tile(hprev + 48 * H_, H_, kk, lane);
      c0 = WMMA_BF16(a0, bm, c0);
      c1 = WMMA_BF16(a1, bm, c1);
      c2 = WMMA_BF16(a2, bm, c2);
      c3 = WMMA_BF16(a3, bm, c3);
    }

#pragma unroll
    for (int mt = 0; mt < 4; ++mt) {
      v8f acc = (mt == 0) ? c0 : (mt == 1) ? c1 : (mt == 2) ? c2 : c3;
#pragma unroll
      for (int r = 0; r < 8; ++r) {
        const int bidx = mt * 16 + r + 8 * half;
        float v = acc[r] + pre[(size_t)(bidx * T_ + t) * H_ + jcol];
        v = tanhf(v);
        hnext[bidx * H_ + jcol] = (bf16_t)v;
        const size_t o = (size_t)(bidx * T_ + t) * (2 * H_) + dir * H_ + jcol;
        if (out_f32) out_f32[o] = v;
        else         out_bf16[o] = (bf16_t)v;
      }
    }

    // per-direction global barrier across the NBLK blocks
    __threadfence();
    __syncthreads();
    if (tid == 0) {
      __hip_atomic_fetch_add(myctr, 1u, __ATOMIC_RELEASE, __HIP_MEMORY_SCOPE_AGENT);
      const unsigned target = (unsigned)(NBLK * (s + 1));
      while (__hip_atomic_load(myctr, __ATOMIC_ACQUIRE, __HIP_MEMORY_SCOPE_AGENT) < target) {
        __builtin_amdgcn_s_sleep(1);
      }
    }
    __syncthreads();
  }
}

// ---------------------------------------------------------------------------
extern "C" void kernel_launch(void* const* d_in, const int* in_sizes, int n_in,
                              void* d_out, int out_size, void* d_ws, size_t ws_size,
                              hipStream_t stream) {
  (void)in_sizes; (void)n_in; (void)out_size; (void)ws_size;

  const float* input   = (const float*)d_in[0];
  const float* w_ih_0f = (const float*)d_in[1];
  const float* w_hh_0f = (const float*)d_in[2];
  const float* b_ih_0f = (const float*)d_in[3];
  const float* b_hh_0f = (const float*)d_in[4];
  const float* w_ih_0b = (const float*)d_in[5];
  const float* w_hh_0b = (const float*)d_in[6];
  const float* b_ih_0b = (const float*)d_in[7];
  const float* b_hh_0b = (const float*)d_in[8];
  const float* w_ih_1f = (const float*)d_in[9];
  const float* w_hh_1f = (const float*)d_in[10];
  const float* b_ih_1f = (const float*)d_in[11];
  const float* b_hh_1f = (const float*)d_in[12];
  const float* w_ih_1b = (const float*)d_in[13];
  const float* w_hh_1b = (const float*)d_in[14];
  const float* b_ih_1b = (const float*)d_in[15];
  const float* b_hh_1b = (const float*)d_in[16];

  char* ws = (char*)d_ws;
  size_t off = 0;
  auto alloc = [&](size_t bytes) -> void* {
    void* p = ws + off;
    off = (off + bytes + 255) & ~(size_t)255;
    return p;
  };

  const size_t BT = (size_t)B_ * T_;
  bf16_t* x0   = (bf16_t*)alloc(BT * NIN * 2);        // layer-0 input, bf16
  bf16_t* x1   = (bf16_t*)alloc(BT * 2 * H_ * 2);     // layer-0 output, bf16
  float*  pref = (float*)alloc(BT * H_ * 4);
  float*  preb = (float*)alloc(BT * H_ * 4);
  bf16_t* wih0f = (bf16_t*)alloc((size_t)H_ * NIN * 2);
  bf16_t* wih0b = (bf16_t*)alloc((size_t)H_ * NIN * 2);
  bf16_t* wih1f = (bf16_t*)alloc((size_t)H_ * 2 * H_ * 2);
  bf16_t* wih1b = (bf16_t*)alloc((size_t)H_ * 2 * H_ * 2);
  bf16_t* whh0f = (bf16_t*)alloc((size_t)H_ * H_ * 2);
  bf16_t* whh0b = (bf16_t*)alloc((size_t)H_ * H_ * 2);
  bf16_t* whh1f = (bf16_t*)alloc((size_t)H_ * H_ * 2);
  bf16_t* whh1b = (bf16_t*)alloc((size_t)H_ * H_ * 2);
  bf16_t* hbuf  = (bf16_t*)alloc((size_t)2 * 2 * B_ * H_ * 2);
  unsigned* ctr = (unsigned*)alloc(256);

  // ---- convert activations + weights to bf16 ----
  cvt_f32_to_bf16<<<2048, 256, 0, stream>>>(input,   x0,    (int)(BT * NIN));
  cvt_f32_to_bf16<<<256,  256, 0, stream>>>(w_ih_0f, wih0f, H_ * NIN);
  cvt_f32_to_bf16<<<256,  256, 0, stream>>>(w_ih_0b, wih0b, H_ * NIN);
  cvt_f32_to_bf16<<<256,  256, 0, stream>>>(w_ih_1f, wih1f, H_ * 2 * H_);
  cvt_f32_to_bf16<<<256,  256, 0, stream>>>(w_ih_1b, wih1b, H_ * 2 * H_);
  cvt_f32_to_bf16<<<256,  256, 0, stream>>>(w_hh_0f, whh0f, H_ * H_);
  cvt_f32_to_bf16<<<256,  256, 0, stream>>>(w_hh_0b, whh0b, H_ * H_);
  cvt_f32_to_bf16<<<256,  256, 0, stream>>>(w_hh_1f, whh1f, H_ * H_);
  cvt_f32_to_bf16<<<256,  256, 0, stream>>>(w_hh_1b, whh1b, H_ * H_);

  (void)hipFuncSetAttribute((const void*)rnn_scan_kernel,
                            hipFuncAttributeMaxDynamicSharedMemorySize, LDS_BYTES);

  const dim3 gemm_grid((unsigned)(BT / 128), H_ / 64, 2);

  // ---- layer 0 ----
  gemm_pre_kernel<<<gemm_grid, 256, 0, stream>>>(x0, NIN, wih0f, wih0b,
                                                 b_ih_0f, b_hh_0f, b_ih_0b, b_hh_0b,
                                                 pref, preb);
  init_state<<<256, 256, 0, stream>>>(hbuf, 2 * 2 * B_ * H_, ctr, 8);
  rnn_scan_kernel<<<dim3(NBLK, 2), 256, LDS_BYTES, stream>>>(
      pref, preb, whh0f, whh0b, hbuf, ctr, x1, nullptr);

  // ---- layer 1 ----
  gemm_pre_kernel<<<gemm_grid, 256, 0, stream>>>(x1, 2 * H_, wih1f, wih1b,
                                                 b_ih_1f, b_hh_1f, b_ih_1b, b_hh_1b,
                                                 pref, preb);
  init_state<<<256, 256, 0, stream>>>(hbuf, 2 * 2 * B_ * H_, ctr, 8);
  rnn_scan_kernel<<<dim3(NBLK, 2), 256, LDS_BYTES, stream>>>(
      pref, preb, whh1f, whh1b, hbuf, ctr, nullptr, (float*)d_out);
}
